// Bert_39977555591829
// MI455X (gfx1250) — matrix operand order repaired and checked by reading, not compiled, not run on hardware
//
#include <hip/hip_runtime.h>
#include <math.h>

// ---------------------------------------------------------------------------
// BERT-base forward for gfx1250 (MI455X).  All GEMMs run on the WMMA pipe
// (v_wmma_f32_16x16x32_bf16), fp32 -> bf16 staged through double-buffered LDS
// (one barrier per K-step, staging of tile t+1 overlaps WMMAs of tile t).
// Block tile 128x128 (8 wave32 waves), each wave owns a 64x32 tile
// (8 accumulators -> 8 WMMAs per K-step).  Branch-free staging (clamp+select)
// keeps EXEC all-ones around the WMMAs; tile t+2 prefetched with
// global_prefetch_b8.
// ---------------------------------------------------------------------------

#define BB 8
#define SS 512
#define DD 768
#define HH 12
#define LL 12
#define FF 3072
#define DHD 64

#define TM 128
#define TN 128
#define TK 32

typedef __attribute__((ext_vector_type(16))) __bf16 v16bf;
typedef __attribute__((ext_vector_type(8)))  float  v8f;

union Frag { v16bf v; unsigned int u[8]; };

// fp32 -> bf16 bit pattern, round-to-nearest-even (fallback path)
__device__ __forceinline__ unsigned short f2bf(float x) {
    union { float f; unsigned int u; } c; c.f = x;
    unsigned int r = (c.u + 0x7fffu + ((c.u >> 16) & 1u)) >> 16;
    return (unsigned short)r;
}

// pack two fp32 -> packed bf16x2 (HW v_cvt_pk_bf16_f32 when available)
__device__ __forceinline__ unsigned int pack2bf(float x, float y) {
#if __has_builtin(__builtin_amdgcn_cvt_pk_bf16_f32)
    typedef __attribute__((ext_vector_type(2))) __bf16 v2bf;
    union { v2bf v; unsigned int u; } c;
    c.v = __builtin_amdgcn_cvt_pk_bf16_f32(x, y);
    return c.u;
#else
    return (unsigned int)f2bf(x) | ((unsigned int)f2bf(y) << 16);
#endif
}

// pack float4 -> 4 bf16, single 64-bit LDS store
__device__ __forceinline__ void store_bf4(unsigned short* dst, float4 v) {
    unsigned long long p = (unsigned long long)pack2bf(v.x, v.y)
                         | ((unsigned long long)pack2bf(v.z, v.w) << 32);
    *reinterpret_cast<unsigned long long*>(dst) = p;
}

// cheap tanh: (e^2x - 1)/(e^2x + 1), clamped for stability
__device__ __forceinline__ float fast_tanh(float x) {
    x = fminf(fmaxf(x, -15.f), 15.f);
    float e = __expf(2.0f * x);
    return (e - 1.0f) / (e + 1.0f);
}

// ---- stage one 128x32 A tile + 32x128/128x32 W tile into LDS (bf16) ----
__device__ __forceinline__ void stage_tile(
    unsigned short (*sA)[TK], unsigned short (*sB)[TK],
    const float* __restrict__ A, int lda, int m0, int M,
    const float* __restrict__ W, int ldw, int opB, int n0, int N,
    int k0, int tid)
{
#pragma unroll
    for (int i = 0; i < 4; ++i) {
        int e  = tid + i * 256;
        int mm = e >> 3;          // 0..127
        int kq = e & 7;           // k = kq*4
        int gm  = m0 + mm;
        int gmc = (gm < M) ? gm : 0;
        float4 v = reinterpret_cast<const float4*>(A + (size_t)gmc * lda + k0)[kq];
        if (gm >= M) { v.x = 0.f; v.y = 0.f; v.z = 0.f; v.w = 0.f; }
        store_bf4(&sA[mm][kq * 4], v);
    }
    if (opB == 0) {
#pragma unroll
        for (int i = 0; i < 4; ++i) {
            int e  = tid + i * 256;
            int kk = e >> 5;      // 0..31
            int nq = e & 31;      // n = nq*4
            int gn  = n0 + nq * 4;
            int gnc = (gn < N) ? gn : 0;   // N is a multiple of 4
            float4 v = *reinterpret_cast<const float4*>(
                           W + (size_t)(k0 + kk) * ldw + gnc);
            if (gn >= N) { v.x = 0.f; v.y = 0.f; v.z = 0.f; v.w = 0.f; }
            sB[nq * 4 + 0][kk] = f2bf(v.x);
            sB[nq * 4 + 1][kk] = f2bf(v.y);
            sB[nq * 4 + 2][kk] = f2bf(v.z);
            sB[nq * 4 + 3][kk] = f2bf(v.w);
        }
    } else {
#pragma unroll
        for (int i = 0; i < 4; ++i) {
            int e  = tid + i * 256;
            int nn = e >> 3;      // 0..127
            int kq = e & 7;
            int gn  = n0 + nn;
            int gnc = (gn < N) ? gn : 0;
            float4 v = reinterpret_cast<const float4*>(
                           W + (size_t)gnc * ldw + k0)[kq];
            if (gn >= N) { v.x = 0.f; v.y = 0.f; v.z = 0.f; v.w = 0.f; }
            store_bf4(&sB[nn][kq * 4], v);
        }
    }
}

// ---------------------------------------------------------------------------
// Generic batched GEMM:  C[z] = act(alpha * A[z] @ W[z] + bias + row_add)
//   A: M x K fp32, row stride lda.  Batch offset = (z/zdiv)*sA1 + (z%zdiv)*sA2
//   W: opB==0 -> K x N row-major (ldw);  opB==1 -> N x K row-major (ldw)
//   act: 0 none, 1 gelu(tanh approx), 2 tanh
//   row_add: optional per-row addend indexed [(z/zdiv)*M + row]  (mask quirk)
// ---------------------------------------------------------------------------
__global__ __launch_bounds__(256)
void gemm_wmma_kernel(
    const float* __restrict__ A, long long sA1, long long sA2, int lda,
    const float* __restrict__ W, long long sW1, long long sW2, int ldw, int opB,
    const float* __restrict__ bias, const float* __restrict__ row_add,
    float* __restrict__ C, long long sC1, long long sC2, int ldc,
    int M, int N, int K, int zdiv, float alpha, int act)
{
    const int z  = blockIdx.z;
    const int zq = z / zdiv;
    const int zr = z - zq * zdiv;
    A += zq * sA1 + zr * sA2;
    W += zq * sW1 + zr * sW2;
    C += zq * sC1 + zr * sC2;

    const int n0 = blockIdx.x * TN;
    const int m0 = blockIdx.y * TM;

    __shared__ unsigned short lA[2][TM][TK];   // [m][k] bf16 bits, ping-pong
    __shared__ unsigned short lB[2][TN][TK];   // [n][k] bf16 bits (B transposed)

    const int tid  = threadIdx.x;
    const int wave = tid >> 5;
    const int lane = tid & 31;
    const int wr   = wave >> 2;   // 0..1  : 64-row block
    const int wc   = wave & 3;    // 0..3  : 32-col block

    const v8f vzero = {0.f, 0.f, 0.f, 0.f, 0.f, 0.f, 0.f, 0.f};
    v8f acc[4][2];
#pragma unroll
    for (int i = 0; i < 4; ++i)
#pragma unroll
        for (int j = 0; j < 2; ++j) acc[i][j] = vzero;

    // prime buffer 0
    stage_tile(lA[0], lB[0], A, lda, m0, M, W, ldw, opB, n0, N, 0, tid);

    int p = 0;
    for (int k0 = 0; k0 < K; k0 += TK) {
        __syncthreads();          // buffer p staged; old reads of p^1 done

        // ---- stage next tile into the other buffer (overlaps WMMAs) ----
        if (k0 + TK < K) {
            stage_tile(lA[p ^ 1], lB[p ^ 1], A, lda, m0, M,
                       W, ldw, opB, n0, N, k0 + TK, tid);
            // prefetch tile t+2 into cache (global_prefetch_b8)
            if (k0 + 2 * TK < K) {
                int pm = m0 + (tid >> 1);
                pm = (pm < M) ? pm : 0;
                __builtin_prefetch(A + (size_t)pm * lda + (k0 + 2 * TK) + (tid & 1) * 16, 0, 3);
                if (opB == 0) {
                    int pk = k0 + 2 * TK + (tid >> 3);
                    __builtin_prefetch(W + (size_t)pk * ldw + n0 + (tid & 7) * 16, 0, 3);
                } else {
                    int pn = n0 + (tid >> 1);
                    pn = (pn < N) ? pn : 0;
                    __builtin_prefetch(W + (size_t)pn * ldw + (k0 + 2 * TK) + (tid & 1) * 16, 0, 3);
                }
            }
        }

        // ---- fragments per ISA 7.12.2 16-bit layouts (buffer p) ----
        Frag a[4], b[2];
        {
            const int lrow = lane & 15;
            const int kb   = (lane >= 16) ? 4 : 0;   // A pair offset
#pragma unroll
            for (int i = 0; i < 4; ++i) {
                const unsigned int* pa =
                    (const unsigned int*)(&lA[p][wr * 64 + i * 16 + lrow][0]);
#pragma unroll
                for (int j = 0; j < 8; ++j) {
                    int kp = (j < 4) ? (j + kb) : (8 + (j - 4) + kb);
                    a[i].u[j] = pa[kp];
                }
            }
            const int kb2 = (lane >= 16) ? 4 : 0;    // B pair offset
#pragma unroll
            for (int j2 = 0; j2 < 2; ++j2) {
                const unsigned int* pb =
                    (const unsigned int*)(&lB[p][wc * 32 + j2 * 16 + lrow][0]);
#pragma unroll
                for (int j = 0; j < 8; ++j) b[j2].u[j] = pb[(j < 4) ? (j + kb2) : (8 + (j - 4) + kb2)];
            }
        }

        // ---- 8 WMMAs per K-step (A-frags reused across both B-frags) ----
#pragma unroll
        for (int i = 0; i < 4; ++i)
#pragma unroll
            for (int j = 0; j < 2; ++j)
                acc[i][j] = __builtin_amdgcn_wmma_f32_16x16x32_bf16(
                    false, a[i].v, false, b[j].v, (short)0, acc[i][j],
                    false, false);

        p ^= 1;
    }

    // ---- epilogue: alpha, bias, row_add (mask quirk), activation ----
    const int colL = lane & 15;
    const int rOff = (lane >= 16) ? 8 : 0;
#pragma unroll
    for (int j = 0; j < 2; ++j) {
        int gcol = n0 + wc * 32 + j * 16 + colL;
        float bv = (bias != nullptr && gcol < N) ? bias[gcol] : 0.f;
#pragma unroll
        for (int i = 0; i < 4; ++i) {
            int rb = m0 + wr * 64 + i * 16 + rOff;
#pragma unroll
            for (int r = 0; r < 8; ++r) {
                int grow = rb + r;
                if (grow < M && gcol < N) {
                    float v = alpha * acc[i][j][r] + bv;
                    if (row_add) v += row_add[(size_t)zq * M + grow];
                    if (act == 1) {
                        float u = v;
                        v = 0.5f * u * (1.0f + fast_tanh(0.7978845608028654f * u *
                                                         (1.0f + 0.044715f * u * u)));
                    } else if (act == 2) {
                        v = fast_tanh(v);
                    }
                    C[(size_t)grow * ldc + gcol] = v;
                }
            }
        }
    }
}

// ---------------------------------------------------------------------------
// Embedding gather: emb = tok_emb[id] + type_emb[tt] + pos_emb[pos]
// ---------------------------------------------------------------------------
__global__ void embed_kernel(const int* __restrict__ ids,
                             const int* __restrict__ tts,
                             const int* __restrict__ pos,
                             const float* __restrict__ tokE,
                             const float* __restrict__ typE,
                             const float* __restrict__ posE,
                             float* __restrict__ out)
{
    int row = blockIdx.x;                // b*S + s
    int s   = row & (SS - 1);
    int id  = ids[row];
    int tt  = tts[row];
    int pd  = pos[s];
#pragma unroll
    for (int j = 0; j < 3; ++j) {
        int d = threadIdx.x + j * 256;
        out[(size_t)row * DD + d] = tokE[(size_t)id * DD + d]
                                  + typE[(size_t)tt * DD + d]
                                  + posE[(size_t)pd * DD + d];
    }
}

// ---------------------------------------------------------------------------
// LayerNorm with optional residual: out = LN(x + res) * g + b   (D = 768)
// ---------------------------------------------------------------------------
__global__ void ln_kernel(const float* __restrict__ x,
                          const float* __restrict__ res,
                          const float* __restrict__ g,
                          const float* __restrict__ b,
                          float* __restrict__ out)
{
    __shared__ float red[256];
    int row = blockIdx.x;
    const float* xr = x + (size_t)row * DD;
    const float* rr = res ? res + (size_t)row * DD : nullptr;
    int tid = threadIdx.x;

    float vals[3];
    float s = 0.f;
#pragma unroll
    for (int j = 0; j < 3; ++j) {
        int d = tid + j * 256;
        float v = xr[d] + (rr ? rr[d] : 0.f);
        vals[j] = v;
        s += v;
    }
    red[tid] = s; __syncthreads();
    for (int off = 128; off > 0; off >>= 1) {
        if (tid < off) red[tid] += red[tid + off];
        __syncthreads();
    }
    float mean = red[0] * (1.0f / DD);
    __syncthreads();

    float vs = 0.f;
#pragma unroll
    for (int j = 0; j < 3; ++j) { float d = vals[j] - mean; vs += d * d; }
    red[tid] = vs; __syncthreads();
    for (int off = 128; off > 0; off >>= 1) {
        if (tid < off) red[tid] += red[tid + off];
        __syncthreads();
    }
    float rstd = rsqrtf(red[0] * (1.0f / DD) + 1e-12f);

#pragma unroll
    for (int j = 0; j < 3; ++j) {
        int d = tid + j * 256;
        out[(size_t)row * DD + d] = (vals[j] - mean) * rstd * g[d] + b[d];
    }
}

// ---------------------------------------------------------------------------
// Row softmax over S = 512 (in place).  One block per row, 2 elems / thread.
// ---------------------------------------------------------------------------
__global__ void softmax_kernel(float* __restrict__ x)
{
    __shared__ float red[256];
    size_t row = blockIdx.x;
    float* xr = x + row * (size_t)SS;
    int tid = threadIdx.x;

    float v0 = xr[tid], v1 = xr[tid + 256];
    red[tid] = fmaxf(v0, v1); __syncthreads();
    for (int off = 128; off > 0; off >>= 1) {
        if (tid < off) red[tid] = fmaxf(red[tid], red[tid + off]);
        __syncthreads();
    }
    float mx = red[0];
    __syncthreads();

    float e0 = __expf(v0 - mx), e1 = __expf(v1 - mx);
    red[tid] = e0 + e1; __syncthreads();
    for (int off = 128; off > 0; off >>= 1) {
        if (tid < off) red[tid] += red[tid + off];
        __syncthreads();
    }
    float inv = 1.0f / red[0];
    xr[tid]       = e0 * inv;
    xr[tid + 256] = e1 * inv;
}

// mask4 precompute: mv[i] = (1 - attention_mask[i]) * -1e9
__global__ void mask_kernel(const float* __restrict__ am, float* __restrict__ mv, int n)
{
    int i = blockIdx.x * blockDim.x + threadIdx.x;
    if (i < n) mv[i] = (1.0f - am[i]) * -1e9f;
}

__global__ void copy_kernel(const float* __restrict__ src, float* __restrict__ dst, int n)
{
    int i = blockIdx.x * blockDim.x + threadIdx.x;
    if (i < n) dst[i] = src[i];
}

// ---------------------------------------------------------------------------
// Host side
// ---------------------------------------------------------------------------
static inline void gemm(hipStream_t st,
                        const float* A, long long sA1, long long sA2, int lda,
                        const float* W, long long sW1, long long sW2, int ldw, int opB,
                        const float* bias, const float* row_add,
                        float* C, long long sC1, long long sC2, int ldc,
                        int M, int N, int K, int Z, int zdiv, float alpha, int act)
{
    dim3 g((N + TN - 1) / TN, (M + TM - 1) / TM, Z);
    gemm_wmma_kernel<<<g, 256, 0, st>>>(A, sA1, sA2, lda,
                                        W, sW1, sW2, ldw, opB,
                                        bias, row_add,
                                        C, sC1, sC2, ldc,
                                        M, N, K, zdiv, alpha, act);
}

extern "C" void kernel_launch(void* const* d_in, const int* in_sizes, int n_in,
                              void* d_out, int out_size, void* d_ws, size_t ws_size,
                              hipStream_t stream)
{
    (void)in_sizes; (void)n_in; (void)out_size; (void)ws_size;

    const int*   input_ids  = (const int*)d_in[0];
    const int*   token_type = (const int*)d_in[1];
    const int*   position   = (const int*)d_in[2];
    const float* attn_mask  = (const float*)d_in[3];
    const float* tok_emb    = (const float*)d_in[4];
    const float* type_emb   = (const float*)d_in[5];
    const float* pos_emb    = (const float*)d_in[6];
    const float* emb_ln_g   = (const float*)d_in[7];
    const float* emb_ln_b   = (const float*)d_in[8];
    const float* qkv_w      = (const float*)d_in[9];
    const float* qkv_b      = (const float*)d_in[10];
    const float* attn_w     = (const float*)d_in[11];
    const float* attn_b     = (const float*)d_in[12];
    const float* ln1_g      = (const float*)d_in[13];
    const float* ln1_b      = (const float*)d_in[14];
    const float* ffn1_w     = (const float*)d_in[15];
    const float* ffn1_b     = (const float*)d_in[16];
    const float* ffn2_w     = (const float*)d_in[17];
    const float* ffn2_b     = (const float*)d_in[18];
    const float* ln2_g      = (const float*)d_in[19];
    const float* ln2_b      = (const float*)d_in[20];
    const float* pool_w     = (const float*)d_in[21];
    const float* pool_b     = (const float*)d_in[22];

    const size_t NT = (size_t)BB * SS;        // 4096 tokens

    // workspace layout (fp32)
    float* ws   = (float*)d_ws;
    float* hbuf = ws;  ws += NT * DD;                       // residual stream
    float* tmp  = ws;  ws += NT * DD;                       // emb / matmul out
    float* h1   = ws;  ws += NT * DD;                       // post-LN1
    float* qkvb = ws;  ws += NT * 3 * DD;                   // fused QKV
    float* ctx  = ws;  ws += NT * DD;                       // attention context
    float* ffnb = ws;  ws += NT * FF;                       // FFN intermediate
    float* mskv = ws;  ws += NT;                            // mask4 values
    float* scor = ws;  ws += (size_t)BB * HH * SS * SS;     // attn logits/probs

    const long long sBH = (long long)SS * 3 * DD;   // qkv batch stride (b)
    const long long sHS = (long long)HH * SS * SS;  // scores stride (b)
    const float qscale = 0.28867513459481287f;      // H**-0.5 (quirk)

    // ---- embeddings + LN ----
    embed_kernel<<<(unsigned)NT, 256, 0, stream>>>(input_ids, token_type, position,
                                                   tok_emb, type_emb, pos_emb, tmp);
    ln_kernel<<<(unsigned)NT, 256, 0, stream>>>(tmp, nullptr, emb_ln_g, emb_ln_b, hbuf);
    mask_kernel<<<((unsigned)NT + 255) / 256, 256, 0, stream>>>(attn_mask, mskv, (int)NT);

    // ---- encoder layers ----
    for (int l = 0; l < LL; ++l) {
        const float* wqkv = qkv_w  + (size_t)l * DD * 3 * DD;
        const float* bqkv = qkv_b  + (size_t)l * 3 * DD;
        const float* wo   = attn_w + (size_t)l * DD * DD;
        const float* bo   = attn_b + (size_t)l * DD;
        const float* g1   = ln1_g  + (size_t)l * DD;
        const float* b1   = ln1_b  + (size_t)l * DD;
        const float* wf1  = ffn1_w + (size_t)l * DD * FF;
        const float* bf1  = ffn1_b + (size_t)l * FF;
        const float* wf2  = ffn2_w + (size_t)l * FF * DD;
        const float* bf2  = ffn2_b + (size_t)l * DD;
        const float* g2   = ln2_g  + (size_t)l * DD;
        const float* b2   = ln2_b  + (size_t)l * DD;

        // QKV projection: [4096 x 768] @ [768 x 2304]
        gemm(stream, hbuf, 0, 0, DD, wqkv, 0, 0, 3 * DD, 0, bqkv, nullptr,
             qkvb, 0, 0, 3 * DD, (int)NT, 3 * DD, DD, 1, 1, 1.0f, 0);

        // scores = (q * H^-0.5) @ k^T  per (b,h):  [512 x 64] @ [64 x 512]
        gemm(stream, qkvb,      sBH, DHD, 3 * DD,
                     qkvb + DD, sBH, DHD, 3 * DD, /*opB=*/1,
             nullptr, nullptr,
             scor, sHS, (long long)SS * SS, SS,
             SS, SS, DHD, BB * HH, HH, qscale, 0);

        softmax_kernel<<<BB * HH * SS, 256, 0, stream>>>(scor);

        // ctx = probs @ v (+ mask4 row add, source quirk): [512 x 512] @ [512 x 64]
        gemm(stream, scor, sHS, (long long)SS * SS, SS,
                     qkvb + 2 * DD, sBH, DHD, 3 * DD, /*opB=*/0,
             nullptr, mskv,
             ctx, (long long)SS * DD, DHD, DD,
             SS, DHD, SS, BB * HH, HH, 1.0f, 0);

        // attention output projection
        gemm(stream, ctx, 0, 0, DD, wo, 0, 0, DD, 0, bo, nullptr,
             tmp, 0, 0, DD, (int)NT, DD, DD, 1, 1, 1.0f, 0);

        ln_kernel<<<(unsigned)NT, 256, 0, stream>>>(tmp, hbuf, g1, b1, h1);

        // FFN1 + GELU
        gemm(stream, h1, 0, 0, DD, wf1, 0, 0, FF, 0, bf1, nullptr,
             ffnb, 0, 0, FF, (int)NT, FF, DD, 1, 1, 1.0f, 1);

        // FFN2
        gemm(stream, ffnb, 0, 0, FF, wf2, 0, 0, DD, 0, bf2, nullptr,
             tmp, 0, 0, DD, (int)NT, DD, FF, 1, 1, 1.0f, 0);

        ln_kernel<<<(unsigned)NT, 256, 0, stream>>>(tmp, h1, g2, b2, hbuf);
    }

    // ---- outputs: h (flat) then pooled ----
    float* out = (float*)d_out;
    int nh = (int)(NT * DD);
    copy_kernel<<<(nh + 255) / 256, 256, 0, stream>>>(hbuf, out, nh);

    // pooled = tanh(h[:,0,:] @ pool_w + pool_b): M=8, lda = S*D
    gemm(stream, hbuf, 0, 0, SS * DD, pool_w, 0, 0, DD, 0, pool_b, nullptr,
         out + NT * DD, 0, 0, DD, BB, DD, DD, 1, 1, 1.0f, 2);
}
